// ARNet_22359599743051
// MI455X (gfx1250) — matrix operand order repaired and checked by reading, not compiled, not run on hardware
//
#include <hip/hip_runtime.h>
#include <math.h>

typedef __attribute__((ext_vector_type(16))) _Float16 v16h;
typedef __attribute__((ext_vector_type(8)))  float    v8f;

#define B_DIM 64
#define N_DIM 256
#define MD    32
#define CH    128
#define CLAMP_V 2.0f
#define EPS_V 1e-8f

__device__ __forceinline__ float fast_rcp(float x)  { return __builtin_amdgcn_rcpf(x); }
__device__ __forceinline__ float fast_sqrt(float x) { return __builtin_amdgcn_sqrtf(x); }

__device__ __forceinline__ float sigmoid_f(float x) {
    // v_exp_f32 + v_rcp_f32 (no IEEE divide expansion)
    return fast_rcp(1.0f + __expf(-x));
}
__device__ __forceinline__ float lipswish_f(float x) {
    // 0.909 * x * sigmoid(x)
    return 0.909f * x * sigmoid_f(x);
}

__global__ __launch_bounds__(256) void egnn_wmma_kernel(
    const float* __restrict__ x,   const float* __restrict__ mask,
    const float* __restrict__ W1,  const float* __restrict__ b1,
    const float* __restrict__ W2,  const float* __restrict__ b2,
    const float* __restrict__ Wg,  const float* __restrict__ bg,
    const float* __restrict__ Wc1, const float* __restrict__ bc1,
    const float* __restrict__ Wc2, const float* __restrict__ bc2,
    const float* __restrict__ scale, float* __restrict__ out)
{
    __shared__ float smem[8][16 * 32];          // per-wave D->A transpose staging
    const int lane = threadIdx.x & 31;
    const int wid  = threadIdx.x >> 5;
    float* med = smem[wid];
    const bool hi  = lane >= 16;
    const int  col = lane & 15;                 // lane's edge row M and B-matrix column

    // ---------------- preload weights into WMMA layouts (once per wave) ----------------
    // m1 A-matrix K map: lanes0-15 K={0..15}, lanes16-31 K={16..31}
    float w1v[16], b1v[16];
    #pragma unroll
    for (int t = 0; t < 16; ++t) {
        int kk = t + (hi ? 16 : 0);
        w1v[t] = W1[kk];
        b1v[t] = b1[kk];
    }
    // B layout (32x16 f16): column = lane%16, K = t + 16*(lane>=16)
    v16h bw2a, bw2b;
    #pragma unroll
    for (int t = 0; t < 16; ++t) {
        int K = t + (hi ? 16 : 0);
        bw2a[t] = (_Float16)W2[K * MD + col];
        bw2b[t] = (_Float16)W2[K * MD + col + 16];
    }
    const float b2a = b2[col], b2b = b2[col + 16];
    const float wga = Wg[col], wgb = Wg[col + 16];
    const float bgv = bg[0], bc2v = bc2[0], scl = scale[0];

    v16h  bwc1[8];
    float bc1v[8], wc2v[8];
    #pragma unroll
    for (int cg = 0; cg < 8; ++cg) {
        int n = cg * 16 + col;
        #pragma unroll
        for (int t = 0; t < 16; ++t) {
            int K = t + (hi ? 16 : 0);
            bwc1[cg][t] = (_Float16)Wc1[K * CH + n];
        }
        bc1v[cg] = bc1[n];
        wc2v[cg] = Wc2[n];
    }

    // ---------------- each wave owns one (b,i) row: 16 tiles of 16 j's ----------------
    const int nwaves = gridDim.x * (blockDim.x >> 5);
    const int wave_gid = blockIdx.x * (blockDim.x >> 5) + wid;
    const int total_rows = B_DIM * N_DIM;

    for (int row = wave_gid; row < total_rows; row += nwaves) {
        const int b = row / N_DIM;
        const int i = row % N_DIM;
        const long bi = (long)b * N_DIM + i;
        const float xi0 = x[bi * 3 + 0], xi1 = x[bi * 3 + 1], xi2 = x[bi * 3 + 2];

        float acc0 = 0.f, acc1 = 0.f, acc2 = 0.f;
        v8f zero = {};

        for (int jt = 0; jt < N_DIM / 16; ++jt) {
            const int j = jt * 16 + col;          // this lane's neighbor (dup across halves)
            const long bj = (long)b * N_DIM + j;
            const float dx0 = xi0 - x[bj * 3 + 0];
            const float dx1 = xi1 - x[bj * 3 + 1];
            const float dx2 = xi2 - x[bj * 3 + 2];
            const float d = 2.0f * (dx0 * dx0 + dx1 * dx1 + dx2 * dx2); // 6-D rel_dist

            // stage 1: m1 = lipswish(d*W1+b1), built directly in A layout
            v16h a1;
            #pragma unroll
            for (int t = 0; t < 16; ++t)
                a1[t] = (_Float16)lipswish_f(fmaf(d, w1v[t], b1v[t]));

            // stage 2: m2 = m1 @ W2  (two WMMAs for 32 output cols)
            v8f d0 = __builtin_amdgcn_wmma_f32_16x16x32_f16(false, a1, false, bw2a,
                                                            (short)0, zero, false, false);
            v8f d1 = __builtin_amdgcn_wmma_f32_16x16x32_f16(false, a1, false, bw2b,
                                                            (short)0, zero, false, false);

            float mA[8], mB[8], p[8];
            #pragma unroll
            for (int r = 0; r < 8; ++r) {
                mA[r] = lipswish_f(d0[r] + b2a);
                mB[r] = lipswish_f(d1[r] + b2b);
                p[r]  = mA[r] * wga + mB[r] * wgb; // gate partial over this lane's 2 cols
            }
            // soft-edge gate: reduce over the 16 feature-lanes of each half-wave
            #pragma unroll
            for (int r = 0; r < 8; ++r) {
                float v = p[r];
                v += __shfl_xor(v, 1);
                v += __shfl_xor(v, 2);
                v += __shfl_xor(v, 4);
                v += __shfl_xor(v, 8);
                const float g = sigmoid_f(v + bgv);
                mA[r] *= g;
                mB[r] *= g;
            }

            // D layout -> A layout transpose via wave-private LDS
            __builtin_amdgcn_wave_barrier();
            #pragma unroll
            for (int r = 0; r < 8; ++r) {
                const int rr = r + (hi ? 8 : 0);     // edge row M
                med[rr * 32 + col]      = mA[r];
                med[rr * 32 + col + 16] = mB[r];
            }
            __builtin_amdgcn_wave_barrier();
            v16h a2;
            {
                const int base1 = hi ? 16 : 0;
                const int base2 = hi ? 24 : 8;
                #pragma unroll
                for (int t = 0; t < 8; ++t) a2[t]     = (_Float16)med[col * 32 + base1 + t];
                #pragma unroll
                for (int t = 0; t < 8; ++t) a2[8 + t] = (_Float16)med[col * 32 + base2 + t];
            }
            __builtin_amdgcn_wave_barrier();

            // stage 3: h = lipswish(m @ Wc1 + bc1); partial w = h @ Wc2 (8 WMMAs)
            float part[8];
            #pragma unroll
            for (int r = 0; r < 8; ++r) part[r] = 0.f;
            #pragma unroll
            for (int cg = 0; cg < 8; ++cg) {
                v8f dh = __builtin_amdgcn_wmma_f32_16x16x32_f16(false, a2, false, bwc1[cg],
                                                                (short)0, zero, false, false);
                #pragma unroll
                for (int r = 0; r < 8; ++r)
                    part[r] = fmaf(lipswish_f(dh[r] + bc1v[cg]), wc2v[cg], part[r]);
            }
            // reduce 128-feature dot across half-wave lanes; clamp
            float s[8];
            #pragma unroll
            for (int r = 0; r < 8; ++r) {
                float v = part[r];
                v += __shfl_xor(v, 1);
                v += __shfl_xor(v, 2);
                v += __shfl_xor(v, 4);
                v += __shfl_xor(v, 8);
                v += bc2v;
                s[r] = fminf(fmaxf(v, -CLAMP_V), CLAMP_V);
            }
            // redistribute so lane gets w for its edge m = lane%16
            const int rsel = col & 7;
            float wv = s[0];
            #pragma unroll
            for (int r = 1; r < 8; ++r) wv = (rsel == r) ? s[r] : wv;
            const float wsw = __shfl_xor(wv, 24);
            const bool need = ((((lane >> 3) ^ (lane >> 4)) & 1) != 0);
            wv = need ? wsw : wv;

            wv *= mask[bj];                       // mask_j (mask_i applied at the end)

            // CoorsNorm: 6-D norm = sqrt(d); both 3-D halves identical
            const float inv = scl * fast_rcp(fmaxf(fast_sqrt(d), EPS_V));
            acc0 = fmaf(wv * dx0, inv, acc0);
            acc1 = fmaf(wv * dx1, inv, acc1);
            acc2 = fmaf(wv * dx2, inv, acc2);
        }

        // halves computed identical contributions -> keep lower half only, reduce 32 lanes
        if (hi) { acc0 = 0.f; acc1 = 0.f; acc2 = 0.f; }
        #pragma unroll
        for (int off = 1; off < 32; off <<= 1) {
            acc0 += __shfl_xor(acc0, off);
            acc1 += __shfl_xor(acc1, off);
            acc2 += __shfl_xor(acc2, off);
        }
        if (lane == 0) {
            const float mi = mask[bi];
            const float o0 = (xi0 + mi * acc0) * mi;
            const float o1 = (xi1 + mi * acc1) * mi;
            const float o2 = (xi2 + mi * acc2) * mi;
            float* op = out + bi * 6;
            op[0] = o0; op[1] = o1; op[2] = o2;   // coors = concat(x,x): duplicate halves
            op[3] = o0; op[4] = o1; op[5] = o2;
        }
    }
}

extern "C" void kernel_launch(void* const* d_in, const int* in_sizes, int n_in,
                              void* d_out, int out_size, void* d_ws, size_t ws_size,
                              hipStream_t stream) {
    (void)in_sizes; (void)n_in; (void)out_size; (void)d_ws; (void)ws_size;
    const float* x    = (const float*)d_in[0];
    const float* mask = (const float*)d_in[1];
    const float* W1   = (const float*)d_in[2];
    const float* b1   = (const float*)d_in[3];
    const float* W2   = (const float*)d_in[4];
    const float* b2   = (const float*)d_in[5];
    const float* Wg   = (const float*)d_in[6];
    const float* bg   = (const float*)d_in[7];
    const float* Wc1  = (const float*)d_in[8];
    const float* bc1  = (const float*)d_in[9];
    const float* Wc2  = (const float*)d_in[10];
    const float* bc2  = (const float*)d_in[11];
    const float* scal = (const float*)d_in[12];
    float* out = (float*)d_out;

    // 16384 waves = one wave per (b,i) row; 2048 blocks x 256 threads
    egnn_wmma_kernel<<<dim3(2048), dim3(256), 0, stream>>>(
        x, mask, W1, b1, W2, b2, Wg, bg, Wc1, bc1, Wc2, bc2, scal, out);
}
